// GaussianDiffusionSampler_61117384622700
// MI455X (gfx1250) — matrix operand order repaired and checked by reading, not compiled, not run on hardware
//
#include <hip/hip_runtime.h>
#include <hip/hip_bf16.h>

#define HID 64
#define T_STEPS 1000
#define ROWS_PER_BLOCK 64
#define MAIN_THREADS 128   // 4 waves of 32; each wave owns 16 rows

typedef __attribute__((ext_vector_type(16))) __bf16         v16bf;
typedef __attribute__((ext_vector_type(8)))  float          v8f;
typedef __attribute__((ext_vector_type(8)))  unsigned short v8u;

union ABu { v8u h[2]; v16bf v; };

__device__ __forceinline__ unsigned short f2bf(float f) {
    unsigned u = __float_as_uint(f);
    unsigned r = ((u >> 16) & 1u) + 0x7fffu;   // round to nearest even
    return (unsigned short)((u + r) >> 16);
}
__device__ __forceinline__ float bf2f(unsigned short s) {
    return __uint_as_float(((unsigned)s) << 16);
}
__device__ __forceinline__ unsigned hash_u32(unsigned v) {
    v ^= v >> 16; v *= 0x7feb352dU;
    v ^= v >> 15; v *= 0x846ca68bU;
    v ^= v >> 16; return v;
}

// ---------------------------------------------------------------------------
// Kernel A: diffusion coefficient tables (sequential cumprod; trivial cost)
// ---------------------------------------------------------------------------
__global__ void coeff_kernel(float* __restrict__ c1mc2, float* __restrict__ sv) {
    if (threadIdx.x != 0) return;
    float abar = 1.0f;
    for (int i = 0; i < T_STEPS; ++i) {
        float beta  = 1e-4f + (0.02f - 1e-4f) * (float)i / (float)(T_STEPS - 1);
        float alpha = 1.0f - beta;
        float abar_prev = abar;
        abar *= alpha;
        float c1 = __fsqrt_rn(1.0f / alpha);
        float c2 = c1 * beta / __fsqrt_rn(1.0f - abar);
        c1mc2[i] = c1 - c2;
        if (i == 1) {
            // posterior_var[1] = beta1 * (1 - abar[0]) / (1 - abar[1])
            sv[0] = __fsqrt_rn(beta * (1.0f - abar_prev) / (1.0f - abar));
        }
        if (i >= 1) sv[i] = __fsqrt_rn(beta);
    }
}

// ---------------------------------------------------------------------------
// Kernel B: temb table [T_STEPS x HID] — the step MLP is batch-invariant.
// One block per time step.
// ---------------------------------------------------------------------------
__global__ void __launch_bounds__(128)
temb_kernel(const float* __restrict__ W1, const float* __restrict__ b1,
            const float* __restrict__ W2, const float* __restrict__ b2,
            float* __restrict__ temb_tab) {
    __shared__ float tin[HID];
    __shared__ float hbuf[2 * HID];
    const int t   = blockIdx.x;
    const int tid = threadIdx.x;

    if (tid < 32) {
        float freq = expf(-logf(10000.0f) * (float)tid / 31.0f);
        float ang  = (float)t * freq;
        tin[tid]      = sinf(ang);
        tin[tid + 32] = cosf(ang);
    }
    __syncthreads();
    {   // h = gelu(tin @ W1 + b1), tanh approximation (jax.nn.gelu default)
        float s = b1[tid];
        #pragma unroll 8
        for (int k = 0; k < HID; ++k) s += tin[k] * W1[k * 128 + tid];
        float g = 0.5f * s * (1.0f + tanhf(0.7978845608028654f * (s + 0.044715f * s * s * s)));
        hbuf[tid] = g;
    }
    __syncthreads();
    if (tid < HID) {
        float o = b2[tid];
        #pragma unroll 8
        for (int j = 0; j < 128; ++j) o += hbuf[j] * W2[j * HID + tid];
        temb_tab[t * HID + tid] = o;
    }
}

// ---------------------------------------------------------------------------
// Kernel C: persistent per-row-tile sampler. 64 rows stay in LDS (fp32) for
// all 1000 steps; per step: packed elementwise scale -> bf16 A tile, WMMA
// against hoisted diff_W fragments, scale/bias/paired-noise, write back.
// ---------------------------------------------------------------------------
__global__ void __launch_bounds__(MAIN_THREADS)
diffusion_main(const float* __restrict__ xT, const float* __restrict__ fg,
               const float* __restrict__ W,  const float* __restrict__ bvec,
               const float* __restrict__ temb_tab,
               const float* __restrict__ c1mc2, const float* __restrict__ sv,
               float* __restrict__ out) {
    __shared__ float          x_lds[ROWS_PER_BLOCK * HID];   // 16 KB, fp32 state
    __shared__ unsigned short fg_lds[ROWS_PER_BLOCK * HID];  //  8 KB, bf16 guide
    __shared__ unsigned short A_lds[ROWS_PER_BLOCK * HID];   //  8 KB, bf16 activations
    __shared__ unsigned short Wt_lds[HID * HID];             //  8 KB, diff_W^T bf16
    __shared__ float          temb_lds[2][HID];              // double-buffered
    __shared__ float          bias_lds[HID];

    const int    tid     = threadIdx.x;
    const int    rowbase = blockIdx.x * ROWS_PER_BLOCK;
    const size_t gbase   = (size_t)rowbase * HID;

    for (int i = tid; i < ROWS_PER_BLOCK * HID; i += MAIN_THREADS) {
        x_lds[i]  = xT[gbase + i];
        fg_lds[i] = f2bf(fg[gbase + i]);
    }
    for (int i = tid; i < HID * HID; i += MAIN_THREADS) {
        int k = i >> 6, n = i & 63;                 // W is [K=64][N=64] row-major
        Wt_lds[n * HID + k] = f2bf(W[i]);           // store column-major (K contiguous)
    }
    if (tid < HID) {
        bias_lds[tid] = bvec[tid];
        temb_lds[(T_STEPS - 1) & 1][tid] = temb_tab[(T_STEPS - 1) * HID + tid];
    }
    __syncthreads();

    const int wave = tid >> 5;
    const int lane = tid & 31;
    // A-fragment addressing (16-bit A 16x32, ISA 7.12.2)
    const int arow = (wave << 4) + (lane & 15);
    const int akb  = (lane < 16) ? 0 : 8;
    // B-fragment addressing (16-bit B 32x16)
    const int bn   = lane & 15;
    const int bkb  = (lane < 16) ? 0 : 8;
    const int radd = (lane < 16) ? 0 : 8;           // C/D row shift

    // diff_W is loop-invariant: hoist all 8 B fragments into registers.
    ABu bfrag[2][4];
    #pragma unroll
    for (int kk = 0; kk < 2; ++kk)
        #pragma unroll
        for (int nt = 0; nt < 4; ++nt) {
            const v8u* p = (const v8u*)&Wt_lds[(nt * 16 + bn) * HID + kk * 32 + bkb];
            bfrag[kk][nt].h[0] = p[0];   // K = kb+0..kb+7
            bfrag[kk][nt].h[1] = p[2];   // K = kb+16..kb+23
        }
    // bias per owned column, hoisted
    float bias_r[4];
    #pragma unroll
    for (int nt = 0; nt < 4; ++nt) bias_r[nt] = bias_lds[nt * 16 + bn];

    // step-invariant noise seeds: one per output row-pair (4 nt x 4 pairs)
    unsigned gm[16];
    #pragma unroll
    for (int nt = 0; nt < 4; ++nt)
        #pragma unroll
        for (int rp = 0; rp < 4; ++rp) {
            int rl = (wave << 4) + rp * 2 + radd;
            unsigned gidx = (unsigned)(rowbase + rl) * (unsigned)HID
                          + (unsigned)(nt * 16 + bn);
            gm[nt * 4 + rp] = gidx * 0x9E3779B9u;
        }

    for (int t = T_STEPS - 1; t >= 0; --t) {
        const float* tl = temb_lds[t & 1];
        // phase 1 (packed x2): A = bf16(x * temb[t] * fg)
        for (int i = tid * 2; i < ROWS_PER_BLOCK * HID; i += MAIN_THREADS * 2) {
            float2   x2  = *(const float2*)&x_lds[i];
            unsigned fgp = *(const unsigned*)&fg_lds[i];
            int      k0  = i & 63;                         // i even -> k0 <= 62
            float a0 = x2.x * tl[k0]     * bf2f((unsigned short)(fgp & 0xffffu));
            float a1 = x2.y * tl[k0 + 1] * bf2f((unsigned short)(fgp >> 16));
            unsigned pk = (unsigned)f2bf(a0) | ((unsigned)f2bf(a1) << 16);
            *(unsigned*)&A_lds[i] = pk;
        }
        __syncthreads();

        const float    scale = c1mc2[t];
        const float    svt   = (t > 0) ? sv[t] : 0.0f;
        const unsigned tA    = (unsigned)t;                       // uniform
        const unsigned tB    = tA * 0x85EBCA6Bu + 0x6A09E667u;    // uniform

        // phase 2: load A fragments per documented striping
        ABu a0f, a1f;
        {
            const v8u* p0 = (const v8u*)&A_lds[arow * HID + 0 + akb];
            a0f.h[0] = p0[0];        // K = akb+0..7
            a0f.h[1] = p0[2];        // K = akb+16..23  (+16 halfs)
            const v8u* p1 = (const v8u*)&A_lds[arow * HID + 32 + akb];
            a1f.h[0] = p1[0];
            a1f.h[1] = p1[2];
        }
        #pragma unroll
        for (int nt = 0; nt < 4; ++nt) {
            v8f c = {};
            c = __builtin_amdgcn_wmma_f32_16x16x32_bf16(false, a0f.v, false, bfrag[0][nt].v,
                                                        (short)0, c, false, false);
            c = __builtin_amdgcn_wmma_f32_16x16x32_bf16(false, a1f.v, false, bfrag[1][nt].v,
                                                        (short)0, c, false, false);
            const int   colb = nt * 16 + bn;
            const float sb   = scale * bias_r[nt];          // fold bias into one FMA
            #pragma unroll
            for (int rp = 0; rp < 4; ++rp) {            // row pairs: one Box-Muller each
                unsigned g  = gm[nt * 4 + rp];
                unsigned h1 = hash_u32(g + tA);
                unsigned h2 = hash_u32(h1 ^ (tB + g));
                // -2*ln(u1), u1=(h1+1)*2^-32 in [2^-32,1]: never denormal ->
                // raw v_log_f32 (log2), no fixup:  -2ln(u1)=2ln2*(32-log2(h1+1))
                float lf = __builtin_amdgcn_logf((float)h1 + 1.0f);
                float R  = __builtin_amdgcn_sqrtf(
                               __builtin_fmaf(-1.3862943611f, lf, 44.3614195558f));
                // v_sin/v_cos take revolutions (compute sin(2*pi*x)); u2 is one
                float u2 = (float)h2 * 2.3283064365386963e-10f;
                float sn = __builtin_amdgcn_sinf(u2);
                float cs = __builtin_amdgcn_cosf(u2);
                float sR = svt * R;
                int r  = rp * 2;
                int rl = (wave << 4) + r + radd;
                x_lds[rl * HID + colb]       = __builtin_fmaf(scale, c[r],     sb) + sR * cs;
                x_lds[(rl + 1) * HID + colb] = __builtin_fmaf(scale, c[r + 1], sb) + sR * sn;
            }
        }
        // prefetch next step's temb row into the other buffer (covered by the
        // same closing barrier; phase 1 above read buffer t&1, we write (t-1)&1)
        if (t > 0 && tid < HID)
            temb_lds[(t - 1) & 1][tid] = temb_tab[(t - 1) * HID + tid];
        __syncthreads();   // x_lds + temb prefetch complete, A_lds free
    }

    for (int i = tid; i < ROWS_PER_BLOCK * HID; i += MAIN_THREADS) {
        float v = x_lds[i];
        out[gbase + i] = fminf(1.0f, fmaxf(-1.0f, v));
    }
}

// ---------------------------------------------------------------------------
extern "C" void kernel_launch(void* const* d_in, const int* in_sizes, int n_in,
                              void* d_out, int out_size, void* d_ws, size_t ws_size,
                              hipStream_t stream) {
    const float* xT = (const float*)d_in[0];
    const float* fg = (const float*)d_in[1];
    const float* dW = (const float*)d_in[2];
    const float* db = (const float*)d_in[3];
    const float* W1 = (const float*)d_in[4];
    const float* b1 = (const float*)d_in[5];
    const float* W2 = (const float*)d_in[6];
    const float* b2 = (const float*)d_in[7];

    float* temb_tab = (float*)d_ws;                  // T_STEPS * HID
    float* c1mc2    = temb_tab + T_STEPS * HID;      // T_STEPS
    float* sv       = c1mc2 + T_STEPS;               // T_STEPS

    coeff_kernel<<<1, 32, 0, stream>>>(c1mc2, sv);
    temb_kernel<<<T_STEPS, 128, 0, stream>>>(W1, b1, W2, b2, temb_tab);

    const int B = in_sizes[0] / HID;                 // 16384
    diffusion_main<<<B / ROWS_PER_BLOCK, MAIN_THREADS, 0, stream>>>(
        xT, fg, dW, db, temb_tab, c1mc2, sv, (float*)d_out);
}
